// CausalSelfAttention_86492051407328
// MI455X (gfx1250) — compile-verified, matrix operand order checked by reading, not run on hardware
//
#include <hip/hip_runtime.h>
#include <hip/hip_bf16.h>

#define B_ 4
#define T_ 1024
#define C_ 2048
#define H_ 16
#define D_ 128

typedef __bf16 bf16;
typedef bf16  v16bf __attribute__((ext_vector_type(16)));
typedef bf16  v8bf  __attribute__((ext_vector_type(8)));
typedef float v8f   __attribute__((ext_vector_type(8)));
typedef int   v4i_gv __attribute__((vector_size(16)));   // matches builtin's V4i spelling

// ---------------- async global->LDS copy (gfx1250 ASYNCcnt path) ----------------
#if defined(__has_builtin)
#if __has_builtin(__builtin_amdgcn_global_load_async_to_lds_b128) && \
    __has_builtin(__builtin_amdgcn_s_wait_asynccnt)
#define HAVE_ASYNC_LDS 1
#endif
#endif

__device__ __forceinline__ void copy16_g2lds(const bf16* g, bf16* l) {
#if defined(HAVE_ASYNC_LDS)
  __builtin_amdgcn_global_load_async_to_lds_b128(
      (__attribute__((address_space(1))) v4i_gv*)g,
      (__attribute__((address_space(3))) v4i_gv*)l, 0, 0);
#else
  *(uint4*)l = *(const uint4*)g;
#endif
}

__device__ __forceinline__ void stage_fence() {
#if defined(HAVE_ASYNC_LDS)
  __builtin_amdgcn_s_wait_asynccnt(0);
#endif
  __syncthreads();
}

// ---------------- WMMA fragment helpers (ISA 7.12.2 layouts, wave32) ----------------
__device__ __forceinline__ v16bf load_frag_A(const bf16* p, int ldk) {
  int lane = threadIdx.x & 31;
  int m  = lane & 15;
  int kb = (lane >> 4) * 8;
  v16bf a;
#pragma unroll
  for (int j = 0; j < 8; ++j) {
    int k = ((j < 4) ? 0 : 16) + kb + ((j & 3) * 2);
    a[2 * j]     = p[m * ldk + k];
    a[2 * j + 1] = p[m * ldk + k + 1];
  }
  return a;
}

// element(k,n) = p[k*ldn + n]
__device__ __forceinline__ v16bf load_frag_B_rm(const bf16* p, int ldn) {
  int lane = threadIdx.x & 31;
  int n  = lane & 15;
  int kb = (lane >> 4) * 8;
  v16bf b;
#pragma unroll
  for (int j = 0; j < 8; ++j) {
    int k = ((j < 4) ? 0 : 16) + kb + ((j & 3) * 2);
    b[2 * j]     = p[k * ldn + n];
    b[2 * j + 1] = p[(k + 1) * ldn + n];
  }
  return b;
}

// element(k,n) = p[n*ldk + k]
__device__ __forceinline__ v16bf load_frag_B_t(const bf16* p, int ldk) {
  int lane = threadIdx.x & 31;
  int n  = lane & 15;
  int kb = (lane >> 4) * 8;
  v16bf b;
#pragma unroll
  for (int j = 0; j < 8; ++j) {
    int k = ((j < 4) ? 0 : 16) + kb + ((j & 3) * 2);
    b[2 * j]     = p[n * ldk + k];
    b[2 * j + 1] = p[n * ldk + k + 1];
  }
  return b;
}

#define WMMA_BF16(a, b, c) \
  __builtin_amdgcn_wmma_f32_16x16x32_bf16(false, (a), false, (b), (short)0, (c), false, false)

// ---------------- fp32 -> bf16 bulk convert ----------------
__global__ __launch_bounds__(256) void cvt_bf16_kernel(const float* __restrict__ src,
                                                       bf16* __restrict__ dst, int n) {
  int i = (blockIdx.x * 256 + threadIdx.x) * 8;
  if (i >= n) return;
  float4 a = *(const float4*)(src + i);
  float4 b = *(const float4*)(src + i + 4);
  v8bf o;
  o[0] = (bf16)a.x; o[1] = (bf16)a.y; o[2] = (bf16)a.z; o[3] = (bf16)a.w;
  o[4] = (bf16)b.x; o[5] = (bf16)b.y; o[6] = (bf16)b.z; o[7] = (bf16)b.w;
  *(v8bf*)(dst + i) = o;
}

// ---------------- LoRA rank-16 factor: U[b,t,r] = sum_i X[b,t,i] * dA[b,r,i] ----------------
__global__ __launch_bounds__(128) void lora_u_kernel(const float* __restrict__ X,
                                                     const float* __restrict__ delta,
                                                     int chunk, float* __restrict__ U) {
  int row = blockIdx.x;          // 0 .. B*T-1
  int b   = row >> 10;           // / T_
  int tid = threadIdx.x;
  int r   = tid >> 3;            // 16 r's, 8 lanes each (contiguous within a wave)
  int seg = tid & 7;
  const float* xr = X + (size_t)row * C_;
  const float* ar = delta + (size_t)b * 128 * C_ + (size_t)(chunk * 16 + r) * C_;
  float acc = 0.f;
  for (int i = seg * 4; i < C_; i += 32) {
    float4 xv = *(const float4*)(xr + i);
    float4 av = *(const float4*)(ar + i);
    acc += xv.x * av.x + xv.y * av.y + xv.z * av.z + xv.w * av.w;
  }
  acc += __shfl_xor(acc, 1);
  acc += __shfl_xor(acc, 2);
  acc += __shfl_xor(acc, 4);
  if (seg == 0) U[(size_t)row * 16 + r] = acc;
}

// ---------------- GEMM (bf16 in, BK=64) + LoRA epilogue (+ fused RoPE for q/k) ----------------
// mode 0: q -> rope, *1/sqrt(D), bf16 out | mode 1: k -> rope, bf16 out
// mode 2: v -> bf16 out                   | mode 3: o-proj -> fp32 out
__global__ __launch_bounds__(128) void gemm_lora_kernel(const bf16* __restrict__ Xb,
                                                        const bf16* __restrict__ Wb,
                                                        const float* __restrict__ U,
                                                        const float* __restrict__ delta,
                                                        int dB_chunk, int mode,
                                                        bf16* __restrict__ outb,
                                                        float* __restrict__ outf) {
  __shared__ bf16  Xs[32 * 64];     // 4 KB
  __shared__ bf16  Ws[128 * 64];    // 16 KB
  __shared__ float Yt[32 * 128];    // 16 KB (rope exchange)

  const int tid  = threadIdx.x;
  const int lane = tid & 31;
  const int w    = tid >> 5;
  const int wm   = w & 1;          // 2 waves along M (16 rows each)
  const int wn   = w >> 1;         // 2 waves along N (64 cols each)

  const int m0 = blockIdx.y * 32;        // row in B*T
  const int o0 = blockIdx.x * 128;       // col (one full head)
  const int b  = m0 >> 10;
  const int t0 = m0 & (T_ - 1);

  v8f acc[4];
#pragma unroll
  for (int s = 0; s < 4; ++s) acc[s] = (v8f){0.f, 0.f, 0.f, 0.f, 0.f, 0.f, 0.f, 0.f};

  // ---- main K loop over C, BK = 64 ----
  for (int kk0 = 0; kk0 < C_; kk0 += 64) {
    // X tile: 32 rows x 64 cols = 256 16B chunks, 2/thread
#pragma unroll
    for (int i = 0; i < 2; ++i) {
      int idx = tid + i * 128;
      int row = idx >> 3, cc = (idx & 7) * 8;
      copy16_g2lds(Xb + (size_t)(m0 + row) * C_ + kk0 + cc, &Xs[row * 64 + cc]);
    }
    // W tile: 128 rows x 64 cols = 1024 16B chunks, 8/thread
#pragma unroll
    for (int i = 0; i < 8; ++i) {
      int idx = tid + i * 128;
      int row = idx >> 3, cc = (idx & 7) * 8;
      copy16_g2lds(Wb + (size_t)(o0 + row) * C_ + kk0 + cc, &Ws[row * 64 + cc]);
    }
    if (kk0 + 64 < C_)  // hint next W tile into cache (global_prefetch_b8)
      __builtin_prefetch(Wb + (size_t)(o0 + tid) * C_ + kk0 + 64, 0, 1);
    stage_fence();
#pragma unroll
    for (int kk = 0; kk < 64; kk += 32) {
      v16bf a = load_frag_A(&Xs[wm * 16 * 64 + kk], 64);
#pragma unroll
      for (int s = 0; s < 4; ++s) {
        v16bf bfr = load_frag_B_t(&Ws[(wn * 64 + s * 16) * 64 + kk], 64);
        acc[s] = WMMA_BF16(a, bfr, acc[s]);
      }
    }
    __syncthreads();
  }

  // ---- LoRA epilogue: one K=32 round (r=16 zero-padded), staged at stride 64 ----
  for (int idx = tid; idx < 32 * 32; idx += 128) {
    int row = idx >> 5, k = idx & 31;
    Xs[row * 64 + k] = (k < 16) ? (bf16)U[(size_t)(m0 + row) * 16 + k] : (bf16)0.f;
  }
  for (int idx = tid; idx < 128 * 32; idx += 128) {
    int oo = idx >> 5, k = idx & 31;
    float v = (k < 16)
        ? delta[(size_t)b * 128 * C_ + (size_t)(dB_chunk * 16 + k) * C_ + o0 + oo]
        : 0.f;
    Ws[oo * 64 + k] = (bf16)v;
  }
  __syncthreads();
  {
    v16bf a = load_frag_A(&Xs[wm * 16 * 64], 64);
#pragma unroll
    for (int s = 0; s < 4; ++s) {
      v16bf bfr = load_frag_B_t(&Ws[(wn * 64 + s * 16) * 64], 64);
      acc[s] = WMMA_BF16(a, bfr, acc[s]);
    }
  }
  __syncthreads();

  // ---- epilogue ----
  if (mode >= 2) {  // direct scatter store
#pragma unroll
    for (int s = 0; s < 4; ++s) {
#pragma unroll
      for (int j = 0; j < 8; ++j) {
        int m = wm * 16 + 8 * (lane >> 4) + j;
        int n = wn * 64 + s * 16 + (lane & 15);
        size_t g = (size_t)(m0 + m) * C_ + o0 + n;
        if (mode == 2) outb[g] = (bf16)acc[s][j];
        else           outf[g] = acc[s][j];
      }
    }
    return;
  }

  // q/k: exchange tile through LDS, apply RoPE (d <-> d+-64 pairs stay in-block: BN=128=D)
#pragma unroll
  for (int s = 0; s < 4; ++s) {
#pragma unroll
    for (int j = 0; j < 8; ++j) {
      int m = wm * 16 + 8 * (lane >> 4) + j;
      int n = wn * 64 + s * 16 + (lane & 15);
      Yt[m * 128 + n] = acc[s][j];
    }
  }
  __syncthreads();
  const float qscale = (mode == 0) ? 0.08838834764831845f : 1.0f;  // 1/sqrt(128)
  for (int idx = tid; idx < 32 * 128; idx += 128) {
    int row = idx >> 7, d = idx & 127;
    int t = t0 + row;
    int jf = d & 63;
    float inv  = __powf(10000.f, -(float)(2 * jf) * (1.0f / 128.0f));
    float ang  = (float)t * inv;
    float c    = __cosf(ang), sn = __sinf(ang);
    float v0   = Yt[row * 128 + d];
    float oth  = (d < 64) ? -Yt[row * 128 + d + 64] : Yt[row * 128 + d - 64];
    float val  = (v0 * c + oth * sn) * qscale;
    outb[(size_t)(m0 + row) * C_ + o0 + d] = (bf16)val;
  }
}

// ---------------- Attention: per (b, h, 16-query tile); full score row in LDS ----------------
__global__ __launch_bounds__(128) void attention_kernel(const bf16* __restrict__ Qb,
                                                        const bf16* __restrict__ Kb,
                                                        const bf16* __restrict__ Vb,
                                                        float* __restrict__ Y) {
  extern __shared__ char smem[];
  float* scores = (float*)smem;                                  // 16*1024*4 = 64 KB
  bf16*  probs  = (bf16*)(smem + 65536);                         // 32 KB
  bf16*  qs     = (bf16*)(smem + 65536 + 32768);                 // 4 KB
  bf16*  stage  = (bf16*)(smem + 65536 + 32768 + 4096);          // 16 KB
  float* rowsum = (float*)(smem + 65536 + 32768 + 4096 + 16384); // 64 B

  const int tid  = threadIdx.x;
  const int lane = tid & 31;
  const int w    = tid >> 5;

  const int bid = blockIdx.x;
  const int qt  = bid & 63;            // T/16 query tiles
  const int bh  = bid >> 6;
  const int b   = bh >> 4;
  const int h   = bh & 15;
  const int t0  = qt * 16;
  const size_t headoff = (size_t)h * 128;

  // load Q tile (16 x 128) -> LDS
  for (int idx = tid; idx < 256; idx += 128) {
    int row = idx >> 4, c0 = (idx & 15) * 8;
    copy16_g2lds(Qb + (size_t)(b * T_ + t0 + row) * C_ + headoff + c0, &qs[row * 128 + c0]);
  }
  stage_fence();
  v16bf aq[4];
#pragma unroll
  for (int kk = 0; kk < 4; ++kk) aq[kk] = load_frag_A(&qs[kk * 32], 128);

  // Phase 1: scores = Q * K^T  (stage 64 K-rows per iter; 1 s-tile per wave)
  for (int it = 0; it < 16; ++it) {
    int s0 = it * 64;
    for (int idx = tid; idx < 1024; idx += 128) {
      int row = idx >> 4, c0 = (idx & 15) * 8;
      copy16_g2lds(Kb + (size_t)(b * T_ + s0 + row) * C_ + headoff + c0, &stage[row * 128 + c0]);
    }
    stage_fence();
    v8f sc = (v8f){0.f, 0.f, 0.f, 0.f, 0.f, 0.f, 0.f, 0.f};
#pragma unroll
    for (int kk = 0; kk < 4; ++kk) {
      v16bf bk = load_frag_B_t(&stage[(w * 16) * 128 + kk * 32], 128);
      sc = WMMA_BF16(aq[kk], bk, sc);
    }
#pragma unroll
    for (int j = 0; j < 8; ++j) {
      int m = 8 * (lane >> 4) + j;
      int n = lane & 15;
      scores[m * 1024 + s0 + w * 16 + n] = sc[j];
    }
    __syncthreads();
  }

  // Phase 2: softmax over each 1024-wide row (8 lanes per row)
  {
    int row = tid >> 3, seg = tid & 7;
    float mx = -3.4e38f;
    for (int c = seg; c < 1024; c += 8) mx = fmaxf(mx, scores[row * 1024 + c]);
    mx = fmaxf(mx, __shfl_xor(mx, 1));
    mx = fmaxf(mx, __shfl_xor(mx, 2));
    mx = fmaxf(mx, __shfl_xor(mx, 4));
    float sm = 0.f;
    for (int c = seg; c < 1024; c += 8) {
      float p = __expf(scores[row * 1024 + c] - mx);
      probs[row * 1024 + c] = (bf16)p;
      sm += p;
    }
    sm += __shfl_xor(sm, 1);
    sm += __shfl_xor(sm, 2);
    sm += __shfl_xor(sm, 4);
    if (seg == 0) rowsum[row] = sm;
  }
  __syncthreads();

  // Phase 3: Y = P * V  (stage 32 V-rows per iter; 2 n-tiles per wave)
  const int n0 = w * 32;
  v8f acc0 = (v8f){0.f, 0.f, 0.f, 0.f, 0.f, 0.f, 0.f, 0.f};
  v8f acc1 = (v8f){0.f, 0.f, 0.f, 0.f, 0.f, 0.f, 0.f, 0.f};
  for (int sb = 0; sb < 32; ++sb) {
    int s0 = sb * 32;
    for (int idx = tid; idx < 512; idx += 128) {
      int row = idx >> 4, c0 = (idx & 15) * 8;
      copy16_g2lds(Vb + (size_t)(b * T_ + s0 + row) * C_ + headoff + c0, &stage[row * 128 + c0]);
    }
    stage_fence();
    v16bf ap = load_frag_A(&probs[s0], 1024);
    v16bf b0 = load_frag_B_rm(&stage[n0], 128);
    acc0 = WMMA_BF16(ap, b0, acc0);
    v16bf b1 = load_frag_B_rm(&stage[n0 + 16], 128);
    acc1 = WMMA_BF16(ap, b1, acc1);
    __syncthreads();
  }
#pragma unroll
  for (int j = 0; j < 8; ++j) {
    int m = 8 * (lane >> 4) + j;
    int n = lane & 15;
    float inv = 1.0f / rowsum[m];
    size_t g = (size_t)(b * T_ + t0 + m) * C_ + headoff;
    Y[g + n0 + n]      = acc0[j] * inv;
    Y[g + n0 + 16 + n] = acc1[j] * inv;
  }
}

// ---------------- launch ----------------
extern "C" void kernel_launch(void* const* d_in, const int* in_sizes, int n_in,
                              void* d_out, int out_size, void* d_ws, size_t ws_size,
                              hipStream_t stream) {
  const float* x     = (const float*)d_in[0];
  const float* delta = (const float*)d_in[1];   // (B, 8*16, C)
  const float* Wq    = (const float*)d_in[2];
  const float* Wk    = (const float*)d_in[3];
  const float* Wv    = (const float*)d_in[4];
  const float* Wo    = (const float*)d_in[5];
  float* out = (float*)d_out;

  const size_t BTC = (size_t)B_ * T_ * C_;      // 8,388,608
  const size_t CC  = (size_t)C_ * C_;           // 4,194,304
  char* ws = (char*)d_ws;
  bf16* qb = (bf16*)ws;
  bf16* kb = qb + BTC;
  bf16* vb = kb + BTC;
  bf16* xb = vb + BTC;
  bf16* yb = xb + BTC;
  bf16* wqb = yb + BTC;
  bf16* wkb = wqb + CC;
  bf16* wvb = wkb + CC;
  bf16* wob = wvb + CC;
  float* y  = (float*)(wob + CC);
  float* uq = y + BTC;
  float* uk = uq + (size_t)B_ * T_ * 16;
  float* uv = uk + (size_t)B_ * T_ * 16;
  float* uo = uv + (size_t)B_ * T_ * 16;

  // bulk bf16 conversions (bandwidth-bound, ~96MB total; strips cvt from GEMM loop)
  cvt_bf16_kernel<<<dim3((int)(BTC / 8 / 256)), 256, 0, stream>>>(x, xb, (int)BTC);
  cvt_bf16_kernel<<<dim3((int)(CC / 8 / 256)), 256, 0, stream>>>(Wq, wqb, (int)CC);
  cvt_bf16_kernel<<<dim3((int)(CC / 8 / 256)), 256, 0, stream>>>(Wk, wkb, (int)CC);
  cvt_bf16_kernel<<<dim3((int)(CC / 8 / 256)), 256, 0, stream>>>(Wv, wvb, (int)CC);
  cvt_bf16_kernel<<<dim3((int)(CC / 8 / 256)), 256, 0, stream>>>(Wo, wob, (int)CC);

  dim3 gU(B_ * T_);
  lora_u_kernel<<<gU, 128, 0, stream>>>(x, delta, 0, uq);   // dqA
  lora_u_kernel<<<gU, 128, 0, stream>>>(x, delta, 2, uk);   // dkA
  lora_u_kernel<<<gU, 128, 0, stream>>>(x, delta, 4, uv);   // dvA

  dim3 gG(C_ / 128, (B_ * T_) / 32);
  gemm_lora_kernel<<<gG, 128, 0, stream>>>(xb, wqb, uq, delta, 1, 0, qb, nullptr);
  gemm_lora_kernel<<<gG, 128, 0, stream>>>(xb, wkb, uk, delta, 3, 1, kb, nullptr);
  gemm_lora_kernel<<<gG, 128, 0, stream>>>(xb, wvb, uv, delta, 5, 2, vb, nullptr);

  const size_t shmem = 65536 + 32768 + 4096 + 16384 + 64;   // ~116 KB (< 320 KB WGP LDS)
  (void)hipFuncSetAttribute((const void*)attention_kernel,
                            hipFuncAttributeMaxDynamicSharedMemorySize, (int)shmem);
  attention_kernel<<<dim3(B_ * H_ * (T_ / 16)), 128, shmem, stream>>>(qb, kb, vb, y);

  cvt_bf16_kernel<<<dim3((int)(BTC / 8 / 256)), 256, 0, stream>>>(y, yb, (int)BTC);
  lora_u_kernel<<<gU, 128, 0, stream>>>(y, delta, 6, uo);   // doA on attention output
  gemm_lora_kernel<<<gG, 128, 0, stream>>>(yb, wob, uo, delta, 7, 3, nullptr, out);
}
